// DeformableCrossFrameAttention_83227876262124
// MI455X (gfx1250) — compile-verified, matrix operand order checked by reading, not run on hardware
//
#include <hip/hip_runtime.h>
#include <hip/hip_bf16.h>

typedef __attribute__((ext_vector_type(16))) _Float16 v16h;
typedef __attribute__((ext_vector_type(4)))  _Float16 v4h;
typedef __attribute__((ext_vector_type(8)))  float    v8f;

#define TILE_M 128
#define TILE_N 128
#define TILE_K 32
#define LDSL   40               // padded row length in halves (80B, 16B multiple)

// Problem constants (from reference setup_inputs)
#define BB     8
#define SH     37
#define SW     37
#define NN     (SH * SW)        // 1369
#define CC     1024
#define HEADS  8
#define HD     128
#define NPTS   8
#define MROWS  (BB * NN)        // 10952

// ---------------------------------------------------------------------------
// CDNA5 async copy: global -> LDS, 16 bytes per enabled lane, ASYNCcnt-tracked.
// lds = low 32 bits of the flat shared-memory address (LDS byte offset).
// ---------------------------------------------------------------------------
__device__ __forceinline__ void async_ld_b128(unsigned lds, const void* gptr)
{
    asm volatile("global_load_async_to_lds_b128 %0, %1, off"
                 :: "v"(lds), "v"((unsigned long long)(size_t)gptr)
                 : "memory");
}
__device__ __forceinline__ void wait_async0()
{
    asm volatile("s_wait_asynccnt 0x0" ::: "memory");
}

// ---------------------------------------------------------------------------
// fp32 -> f16 elementwise convert (4 elems / thread)
// ---------------------------------------------------------------------------
__global__ __launch_bounds__(256)
void cvt_f16(const float* __restrict__ in, _Float16* __restrict__ out, long n4)
{
    long i = (long)blockIdx.x * 256 + threadIdx.x;
    if (i >= n4) return;
    float4 v = *(const float4*)(in + i * 4);
    v4h h; h[0] = (_Float16)v.x; h[1] = (_Float16)v.y;
    h[2] = (_Float16)v.z; h[3] = (_Float16)v.w;
    *(v4h*)(out + i * 4) = h;
}

// ---------------------------------------------------------------------------
// W[k][n] (fp32, leading dim ldw) -> Wt[n][k] (f16). K, Ncols multiples of 32.
// ---------------------------------------------------------------------------
__global__ __launch_bounds__(256)
void transpose_cvt(const float* __restrict__ W, _Float16* __restrict__ Wt,
                   int K, int Ncols, int ldw)
{
    __shared__ float tile[32][33];
    const int k0 = blockIdx.x * 32, n0 = blockIdx.y * 32;
    const int tx = threadIdx.x & 31, ty = threadIdx.x >> 5;   // 32 x 8
    #pragma unroll
    for (int s = 0; s < 32; s += 8)
        tile[ty + s][tx] = W[(size_t)(k0 + ty + s) * ldw + n0 + tx];
    __syncthreads();
    #pragma unroll
    for (int s = 0; s < 32; s += 8)
        Wt[(size_t)(n0 + ty + s) * K + k0 + tx] = (_Float16)tile[tx][ty + s];
}

// ---------------------------------------------------------------------------
// GEMM: Y[M x Ncols](fp32) = X[M x K](f16) * Wt[Ncols x K](f16, pre-transposed)
//       + bias, via v_wmma_f32_16x16x32_f16.
// 256 threads = 8 waves; block tile 128x128; wave tile 32x64 (2x4 WMMA).
// Double-buffered LDS filled by global_load_async_to_lds_b128.
// ---------------------------------------------------------------------------
__global__ __launch_bounds__(256)
void gemm_f16_wmma(const _Float16* __restrict__ X, const _Float16* __restrict__ Wt,
                   const float* __restrict__ bias, float* __restrict__ Y,
                   int M, int K, int Ncols)
{
    __shared__ _Float16 As[2][TILE_M][LDSL];
    __shared__ _Float16 Bs[2][TILE_N][LDSL];   // [n][k]

    const int tid  = threadIdx.x;
    const int lane = tid & 31;
    const int wave = tid >> 5;       // 0..7
    const int wm   = wave >> 1;      // 0..3 -> m offset wm*32
    const int wn   = wave & 1;       // 0..1 -> n offset wn*64

    const int m0 = blockIdx.x * TILE_M;
    const int n0 = blockIdx.y * TILE_N;

    v8f acc[2][4] = {};

    const int lrow  = tid >> 2;        // 0..63 (row within pass)
    const int lcol8 = (tid & 3) * 8;   // 0,8,16,24

    const int halfsel = lane >> 4;     // 0 or 1
    const int mrow    = lane & 15;

    // Per-thread source pointers (A rows clamped: OOB rows only feed C rows
    // that the store masks out; WMMA rows are independent).
    const int r0 = lrow, r1 = lrow + 64;
    const _Float16* ga0 = X + (size_t)min(m0 + r0, M - 1) * K + lcol8;
    const _Float16* ga1 = X + (size_t)min(m0 + r1, M - 1) * K + lcol8;
    const _Float16* gb0 = Wt + (size_t)(n0 + r0) * K + lcol8;
    const _Float16* gb1 = Wt + (size_t)(n0 + r1) * K + lcol8;

    // LDS byte addresses (flat shared pointer truncates to LDS offset)
    const unsigned laA0[2] = { (unsigned)(size_t)&As[0][r0][lcol8],
                               (unsigned)(size_t)&As[1][r0][lcol8] };
    const unsigned laA1[2] = { (unsigned)(size_t)&As[0][r1][lcol8],
                               (unsigned)(size_t)&As[1][r1][lcol8] };
    const unsigned laB0[2] = { (unsigned)(size_t)&Bs[0][r0][lcol8],
                               (unsigned)(size_t)&Bs[1][r0][lcol8] };
    const unsigned laB1[2] = { (unsigned)(size_t)&Bs[0][r1][lcol8],
                               (unsigned)(size_t)&Bs[1][r1][lcol8] };

    // prologue: DMA tile 0 into buffer 0
    async_ld_b128(laA0[0], ga0);
    async_ld_b128(laA1[0], ga1);
    async_ld_b128(laB0[0], gb0);
    async_ld_b128(laB1[0], gb1);

    int buf = 0;
    for (int kk = 0; kk < K; kk += TILE_K) {
        wait_async0();        // my DMA for tile kk landed
        __syncthreads();      // everyone's tile kk landed; prev buffer free

        if (kk + TILE_K < K) {
            const int nb = buf ^ 1;
            async_ld_b128(laA0[nb], ga0 + kk + TILE_K);
            async_ld_b128(laA1[nb], ga1 + kk + TILE_K);
            async_ld_b128(laB0[nb], gb0 + kk + TILE_K);
            async_ld_b128(laB1[nb], gb1 + kk + TILE_K);
        }

        // ---- fragments per ISA 16-bit 16x32 layout ----
        v16h afrag[2] = {}, bfrag[4] = {};
        #pragma unroll
        for (int t = 0; t < 2; ++t) {
            const int ar = wm * 32 + t * 16 + mrow;
            #pragma unroll
            for (int v = 0; v < 8; ++v) {
                const int kb = ((v < 4) ? (2 * v) : (16 + 2 * (v - 4))) + halfsel * 8;
                afrag[t][2 * v]     = As[buf][ar][kb];
                afrag[t][2 * v + 1] = As[buf][ar][kb + 1];
            }
        }
        #pragma unroll
        for (int u = 0; u < 4; ++u) {
            const int bn = wn * 64 + u * 16 + mrow;
            #pragma unroll
            for (int v = 0; v < 8; ++v) {
                const int kb = ((v < 4) ? (2 * v) : (16 + 2 * (v - 4))) + halfsel * 8;
                bfrag[u][2 * v]     = Bs[buf][bn][kb];
                bfrag[u][2 * v + 1] = Bs[buf][bn][kb + 1];
            }
        }
        #pragma unroll
        for (int i = 0; i < 2; ++i)
            #pragma unroll
            for (int j = 0; j < 4; ++j)
                acc[i][j] = __builtin_amdgcn_wmma_f32_16x16x32_f16(
                    false, afrag[i], false, bfrag[j],
                    (short)0, acc[i][j], false, false);

        __syncthreads();      // done reading buf before it is DMA'd over next round
        buf ^= 1;
    }

    // ---- store (C layout: VGPR r -> M = r + 8*(lane>>4), N = lane&15) ----
    const int colIn = lane & 15;
    const int rowHi = (lane >> 4) * 8;
    #pragma unroll
    for (int i = 0; i < 2; ++i) {
        #pragma unroll
        for (int j = 0; j < 4; ++j) {
            const int col = n0 + wn * 64 + j * 16 + colIn;
            const float bv = bias[col];
            #pragma unroll
            for (int r = 0; r < 8; ++r) {
                const int row = m0 + wm * 32 + i * 16 + rowHi + r;
                if (row < M)
                    Y[(size_t)row * Ncols + col] = acc[i][j][r] + bv;
            }
        }
    }
}

// ---------------------------------------------------------------------------
// Deformable sampling + softmax attention. One wave32 per (b, n, head).
// Writes f16 output (consumed by the out-proj WMMA GEMM).
// Replicates the reference exactly, incl. x_norm = y_grid + off[...,0].
// ---------------------------------------------------------------------------
__global__ __launch_bounds__(256)
void deform_attn(const float* __restrict__ q, const float* __restrict__ k,
                 const float* __restrict__ off, _Float16* __restrict__ out)
{
    const int lane = threadIdx.x & 31;
    const int wave = threadIdx.x >> 5;
    const long item = (long)blockIdx.x * 8 + wave;     // (b*N + n)*HEADS + h
    if (item >= (long)MROWS * HEADS) return;

    const int  h  = (int)(item & (HEADS - 1));
    const long bn = item >> 3;
    const int  n  = (int)(bn % NN);
    const int  b  = (int)(bn / NN);
    const int  gy = n / SW;
    const int  gx = n % SW;
    const float cy = -1.f + 2.f * (float)gy / (float)(SH - 1);
    const float cx = -1.f + 2.f * (float)gx / (float)(SW - 1);

    const float4 q4 = *(const float4*)(q + bn * CC + h * HD + lane * 4);
    const float* offp = off + bn * (HEADS * NPTS * 2) + h * (NPTS * 2);
    const float* kb   = k + (long)b * NN * CC + h * HD + lane * 4;

    float4 s[NPTS];
    float  sc[NPTS];

    #pragma unroll
    for (int p = 0; p < NPTS; ++p) {
        const float xn = cy + offp[p * 2 + 0];   // reference: x_norm = locs[...,0]
        const float yn = cx + offp[p * 2 + 1];
        float fx = fminf(fmaxf((xn + 1.f) * 0.5f * (float)(SW - 1), 0.f), (float)(SW - 1));
        float fy = fminf(fmaxf((yn + 1.f) * 0.5f * (float)(SH - 1), 0.f), (float)(SH - 1));
        const float x0f = floorf(fx), y0f = floorf(fy);
        const float wx = fx - x0f,    wy = fy - y0f;
        const int x0 = (int)x0f,      y0 = (int)y0f;
        const int x1 = min(x0 + 1, SW - 1);
        const int y1 = min(y0 + 1, SH - 1);

        const float4 v00 = *(const float4*)(kb + (long)(y0 * SW + x0) * CC);
        const float4 v01 = *(const float4*)(kb + (long)(y0 * SW + x1) * CC);
        const float4 v10 = *(const float4*)(kb + (long)(y1 * SW + x0) * CC);
        const float4 v11 = *(const float4*)(kb + (long)(y1 * SW + x1) * CC);

        float4 sv;
        sv.x = (1.f - wy) * ((1.f - wx) * v00.x + wx * v01.x) + wy * ((1.f - wx) * v10.x + wx * v11.x);
        sv.y = (1.f - wy) * ((1.f - wx) * v00.y + wx * v01.y) + wy * ((1.f - wx) * v10.y + wx * v11.y);
        sv.z = (1.f - wy) * ((1.f - wx) * v00.z + wx * v01.z) + wy * ((1.f - wx) * v10.z + wx * v11.z);
        sv.w = (1.f - wy) * ((1.f - wx) * v00.w + wx * v01.w) + wy * ((1.f - wx) * v10.w + wx * v11.w);
        s[p] = sv;

        float d = q4.x * sv.x + q4.y * sv.y + q4.z * sv.z + q4.w * sv.w;
        #pragma unroll
        for (int m = 16; m > 0; m >>= 1) d += __shfl_xor(d, m, 32);
        sc[p] = d * 0.08838834764831845f;   // 1/sqrt(128)
    }

    float mx = sc[0];
    #pragma unroll
    for (int p = 1; p < NPTS; ++p) mx = fmaxf(mx, sc[p]);
    float e[NPTS], sum = 0.f;
    #pragma unroll
    for (int p = 0; p < NPTS; ++p) { e[p] = __expf(sc[p] - mx); sum += e[p]; }
    const float inv = 1.f / sum;

    float4 o; o.x = 0.f; o.y = 0.f; o.z = 0.f; o.w = 0.f;
    #pragma unroll
    for (int p = 0; p < NPTS; ++p) {
        const float a = e[p] * inv;
        o.x += a * s[p].x; o.y += a * s[p].y; o.z += a * s[p].z; o.w += a * s[p].w;
    }
    v4h oh; oh[0] = (_Float16)o.x; oh[1] = (_Float16)o.y;
    oh[2] = (_Float16)o.z; oh[3] = (_Float16)o.w;
    *(v4h*)(out + bn * CC + h * HD + lane * 4) = oh;
}

extern "C" void kernel_launch(void* const* d_in, const int* in_sizes, int n_in,
                              void* d_out, int out_size, void* d_ws, size_t ws_size,
                              hipStream_t stream) {
    const float* query = (const float*)d_in[0];
    const float* ref   = (const float*)d_in[1];
    const float* Wq    = (const float*)d_in[2];
    const float* bq    = (const float*)d_in[3];
    const float* Wkv   = (const float*)d_in[4];
    const float* bkv   = (const float*)d_in[5];
    const float* Woff  = (const float*)d_in[6];
    const float* boff  = (const float*)d_in[7];
    const float* Wout  = (const float*)d_in[8];
    const float* bout  = (const float*)d_in[9];

    const int NOFF = HEADS * NPTS * 2;   // 128

    // ---- workspace carve (all sizes 16B multiples) ----
    char* p = (char*)d_ws;
    float*    qbuf   = (float*)p;    p += (size_t)MROWS * CC * 4;
    float*    kbuf   = (float*)p;    p += (size_t)MROWS * CC * 4;
    float*    offbuf = (float*)p;    p += (size_t)MROWS * NOFF * 4;
    _Float16* qh     = (_Float16*)p; p += (size_t)MROWS * CC * 2;
    _Float16* rh     = (_Float16*)p; p += (size_t)MROWS * CC * 2;
    _Float16* attnh  = (_Float16*)p; p += (size_t)MROWS * CC * 2;
    _Float16* WqT    = (_Float16*)p; p += (size_t)CC * CC * 2;
    _Float16* WkT    = (_Float16*)p; p += (size_t)CC * CC * 2;
    _Float16* WoffT  = (_Float16*)p; p += (size_t)NOFF * CC * 2;
    _Float16* WoutT  = (_Float16*)p; p += (size_t)CC * CC * 2;

    const dim3 blk(256);
    const int mtiles = (MROWS + TILE_M - 1) / TILE_M;   // 86

    // ---- stage 0: one-time f16 conversion / weight transpose ----
    {
        long n4 = (long)MROWS * CC / 4;
        cvt_f16<<<dim3((unsigned)((n4 + 255) / 256)), blk, 0, stream>>>(query, qh, n4);
        cvt_f16<<<dim3((unsigned)((n4 + 255) / 256)), blk, 0, stream>>>(ref,   rh, n4);
        transpose_cvt<<<dim3(CC / 32, CC / 32),   blk, 0, stream>>>(Wq,   WqT,   CC, CC,   CC);
        transpose_cvt<<<dim3(CC / 32, CC / 32),   blk, 0, stream>>>(Wkv,  WkT,   CC, CC,   2 * CC); // k half only
        transpose_cvt<<<dim3(CC / 32, NOFF / 32), blk, 0, stream>>>(Woff, WoffT, CC, NOFF, NOFF);
        transpose_cvt<<<dim3(CC / 32, CC / 32),   blk, 0, stream>>>(Wout, WoutT, CC, CC,   CC);
    }

    // ---- stage 1: projection GEMMs (WMMA, async-DMA double buffered) ----
    gemm_f16_wmma<<<dim3(mtiles, CC / TILE_N), blk, 0, stream>>>(
        qh, WqT, bq, qbuf, MROWS, CC, CC);
    gemm_f16_wmma<<<dim3(mtiles, CC / TILE_N), blk, 0, stream>>>(
        rh, WkT, bkv, kbuf, MROWS, CC, CC);
    gemm_f16_wmma<<<dim3(mtiles, NOFF / TILE_N), blk, 0, stream>>>(
        qh, WoffT, boff, offbuf, MROWS, CC, NOFF);

    // ---- stage 2: deformable sampling + attention ----
    const long items = (long)MROWS * HEADS;             // 87616
    deform_attn<<<dim3((unsigned)((items + 7) / 8)), blk, 0, stream>>>(
        qbuf, kbuf, offbuf, attnh);

    // ---- stage 3: output projection ----
    gemm_f16_wmma<<<dim3(mtiles, CC / TILE_N), blk, 0, stream>>>(
        attnh, WoutT, bout, (float*)d_out, MROWS, CC, CC);
}
